// ReformerLayer_68221260530040
// MI455X (gfx1250) — matrix operand order repaired and checked by reading, not compiled, not run on hardware
//
#include <hip/hip_runtime.h>
#include <hip/hip_bf16.h>
#include <math.h>

typedef _Float16 v16h __attribute__((ext_vector_type(16)));
typedef _Float16 h8   __attribute__((ext_vector_type(8)));
typedef float    v8f  __attribute__((ext_vector_type(8)));

#define C_B   4
#define C_T   4096
#define C_D   512
#define C_HD  8
#define C_DH  64
#define C_NH  4
#define C_NB  64
#define C_BH  32      /* B*HEADS */
#define C_TT  16384   /* N_HASHES * T */
#define C_NCH 256     /* n_chunks */
#define C_FF  2048

// ---------------------------------------------------------------- WMMA utils
__device__ __forceinline__ v8f wmma_f16(v16h a, v16h b, v8f c) {
  return __builtin_amdgcn_wmma_f32_16x16x32_f16(false, a, false, b, (short)0, c,
                                                false, false);
}

// A fragment from a per-lane row pointer (row = M=lane&15 of the tile).
// lane<16: K in {0..7} u {16..23}; lane>=16: K in {8..15} u {24..31}
__device__ __forceinline__ v16h frag_a_row(const _Float16* rowp, int hi) {
  const _Float16* p = rowp + hi * 8;
  h8 lo = *(const h8*)p;
  h8 hg = *(const h8*)(p + 16);
  v16h o;
#pragma unroll
  for (int i = 0; i < 8; ++i) { o[i] = lo[i]; o[i + 8] = hg[i]; }
  return o;
}

// B fragment from a per-lane row pointer into (N,K) storage (row = N=lane&15).
// lane<16: K in {0..7} u {16..23}; lane>=16: K in {8..15} u {24..31}
__device__ __forceinline__ v16h frag_b_row(const _Float16* rowp, int hi) {
  const _Float16* p = rowp + hi * 8;
  h8 lo = *(const h8*)p;
  h8 hg = *(const h8*)(p + 16);
  v16h o;
#pragma unroll
  for (int i = 0; i < 8; ++i) { o[i] = lo[i]; o[i + 8] = hg[i]; }
  return o;
}

// Tile-origin versions (used by the attention kernel, LDS operands)
__device__ __forceinline__ v16h frag_a(const _Float16* tile, int ld) {
  int lane = threadIdx.x & 31;
  return frag_a_row(tile + (size_t)(lane & 15) * ld, lane >> 4);
}
__device__ __forceinline__ v16h frag_b(const _Float16* tile, int ld) {
  int lane = threadIdx.x & 31;
  return frag_b_row(tile + (size_t)(lane & 15) * ld, lane >> 4);
}

// ---------------------------------------------------------------- f32 -> f16
__global__ void cvt_f16_kernel(const float* __restrict__ in,
                               _Float16* __restrict__ out, int n) {
  int i = blockIdx.x * 256 + threadIdx.x;
  if (i < n) out[i] = (_Float16)in[i];
}

// ---------------------------------------------------------------- GEMM C = A @ W^T
// A: (M,K) f16 row-major. Bw: (N,K) f16 row-major (weight layout).
// Register-blocked: 8 waves/block, each wave owns a 64x64 output tile
// (4x4 WMMA tiles), block macro-tile = 128x256.
// mode 0: Cf = C ; mode 1: Cf = C + bias[n] + res ; mode 2: Ch = relu(C + bias[n]) f16
__global__ void __launch_bounds__(256, 1)
gemm_wmma_kernel(const _Float16* __restrict__ A,
                 const _Float16* __restrict__ Bw,
                 float* __restrict__ Cf,
                 _Float16* __restrict__ Ch,
                 const float* __restrict__ bias,
                 const float* __restrict__ res,
                 int M, int N, int K, int mode) {
  int tid = threadIdx.x;
  int lane = tid & 31;
  int wv = tid >> 5;              // 0..7
  int waveM = wv >> 2;            // 0..1
  int waveN = wv & 3;             // 0..3
  int m_base = blockIdx.x * 128 + waveM * 64;
  int n_base = blockIdx.y * 256 + waveN * 64;
  int r = lane & 15;
  int hi = lane >> 4;

  // per-lane row pointers for the 4 A tile-rows and 4 B tile-rows
  const _Float16* aRow[4];
  const _Float16* bRow[4];
#pragma unroll
  for (int mi = 0; mi < 4; ++mi)
    aRow[mi] = A + (size_t)(m_base + mi * 16 + r) * K;
#pragma unroll
  for (int ni = 0; ni < 4; ++ni)
    bRow[ni] = Bw + (size_t)(n_base + ni * 16 + r) * K;

  v8f acc[4][4] = {};
  for (int k0 = 0; k0 < K; k0 += 32) {
    // prefetch two K-steps ahead (global_prefetch_b8; speculative, clamped)
    int kpf = (k0 + 64 < K) ? (k0 + 64) : (K - 32);
#pragma unroll
    for (int mi = 0; mi < 4; ++mi) __builtin_prefetch(aRow[mi] + kpf, 0, 3);
#pragma unroll
    for (int ni = 0; ni < 4; ++ni) __builtin_prefetch(bRow[ni] + kpf, 0, 3);

    v16h af[4];
#pragma unroll
    for (int mi = 0; mi < 4; ++mi) af[mi] = frag_a_row(aRow[mi] + k0, hi);
#pragma unroll
    for (int ni = 0; ni < 4; ++ni) {
      v16h bf = frag_b_row(bRow[ni] + k0, hi);
#pragma unroll
      for (int mi = 0; mi < 4; ++mi)
        acc[mi][ni] = wmma_f16(af[mi], bf, acc[mi][ni]);
    }
  }

#pragma unroll
  for (int mi = 0; mi < 4; ++mi) {
#pragma unroll
    for (int ni = 0; ni < 4; ++ni) {
      int n = n_base + ni * 16 + r;
#pragma unroll
      for (int vr = 0; vr < 8; ++vr) {
        int m = m_base + mi * 16 + vr + hi * 8;
        size_t idx = (size_t)m * N + n;
        float val = acc[mi][ni][vr];
        if (mode == 0) {
          Cf[idx] = val;
        } else if (mode == 1) {
          Cf[idx] = val + bias[n] + res[idx];
        } else {
          val += bias[n];
          Ch[idx] = (_Float16)(val > 0.f ? val : 0.f);
        }
      }
    }
  }
}

// ---------------------------------------------------------------- LSH hashing
// buckets[bh*TT + hash*T + t] = argmax_j([rot,-rot]) + hash*N_BUCKETS
__global__ void hash_kernel(const float* __restrict__ qk,
                            const float* __restrict__ rotations,
                            int* __restrict__ buckets) {
  __shared__ float qs[64 * 64];      // 16 KB: 64 tokens x 64 dims
  __shared__ float rs[64 * 4 * 32];  // 32 KB: rotations (f,h,i)
  int bh = blockIdx.y;
  int b = bh >> 3, hd = bh & 7;
  int t0 = blockIdx.x * 64;
  int tid = threadIdx.x;
  for (int i = tid; i < 64 * 64; i += 256) {
    int row = i >> 6, col = i & 63;
    qs[i] = qk[((size_t)(b * C_T + t0 + row)) * C_D + hd * C_DH + col];
  }
  for (int i = tid; i < 64 * 4 * 32; i += 256) rs[i] = rotations[i];
  __syncthreads();

  int tl = tid & 63;
  int hsh = tid >> 6;
  float s[32];
#pragma unroll
  for (int i = 0; i < 32; ++i) s[i] = 0.f;
  for (int f = 0; f < 64; ++f) {
    float qv = qs[tl * 64 + f];
    const float* rp = &rs[f * 128 + hsh * 32];
#pragma unroll
    for (int i = 0; i < 32; ++i) s[i] += qv * rp[i];
  }
  float best = s[0];
  int arg = 0;
  for (int j = 1; j < 64; ++j) {
    float vj = (j < 32) ? s[j] : -s[j - 32];
    if (vj > best) { best = vj; arg = j; }
  }
  buckets[(size_t)bh * C_TT + hsh * C_T + t0 + tl] = arg + hsh * C_NB;
}

// ---------------------------------------------------------------- histogram + exclusive prefix
__global__ void hist_kernel(const int* __restrict__ buckets,
                            int* __restrict__ offsets) {
  __shared__ int cnt[256];
  int bh = blockIdx.x;
  int tid = threadIdx.x;
  cnt[tid] = 0;
  __syncthreads();
  for (int i = tid; i < C_TT; i += 256)
    atomicAdd(&cnt[buckets[(size_t)bh * C_TT + i]], 1);
  __syncthreads();
  if (tid == 0) {
    int run = 0;
    for (int j = 0; j < 256; ++j) {
      offsets[bh * 256 + j] = run;
      run += cnt[j];
    }
  }
}

// ---------------------------------------------------------------- stable scatter (counting sort)
// sticker[bh][j] = original ticker at sorted position j
__global__ void scatter_kernel(const int* __restrict__ buckets,
                               const int* __restrict__ offsets,
                               int* __restrict__ sticker) {
  __shared__ int scan[256];
  __shared__ int runsh;
  int bin = blockIdx.x;
  int bh = blockIdx.y;
  int tid = threadIdx.x;
  int hsh = bin >> 6;           // bins of one hash round only
  int base = hsh * C_T;
  if (tid == 0) runsh = offsets[bh * 256 + bin];
  __syncthreads();
  for (int it = 0; it < 16; ++it) {
    int idx = it * 256 + tid;
    int tick = base + idx;
    int flag = (buckets[(size_t)bh * C_TT + tick] == bin) ? 1 : 0;
    scan[tid] = flag;
    __syncthreads();
    for (int off = 1; off < 256; off <<= 1) {
      int t2 = (tid >= off) ? scan[tid - off] : 0;
      __syncthreads();
      scan[tid] += t2;
      __syncthreads();
    }
    int runv = runsh;
    if (flag) sticker[(size_t)bh * C_TT + runv + scan[tid] - 1] = tick;
    int tot = scan[255];
    __syncthreads();
    if (tid == 0) runsh = runv + tot;
    __syncthreads();
  }
}

// ---------------------------------------------------------------- chunked LSH attention
// one block per (chunk, bh); 64 queries x 128 keys (look-one-back), dh=64
__global__ void __launch_bounds__(256)
attn_kernel(const float* __restrict__ qk, const float* __restrict__ v,
            const int* __restrict__ sticker, float* __restrict__ obuf,
            float* __restrict__ logits) {
  __shared__ __align__(32) char smem[59392];
  _Float16* q16 = (_Float16*)(smem + 0);       // 64x64, dead after dots
  _Float16* k16 = (_Float16*)(smem + 8192);    // 128x64, dead after dots
  _Float16* vT  = (_Float16*)(smem + 24576);   // 64x128 (dh-major)
  _Float16* d16 = (_Float16*)(smem + 40960);   // 64x128 dots (f16)
  _Float16* p16 = (_Float16*)(smem + 0);       // 64x128 probs, aliases q16/k16
  int*   qt  = (int*)(smem + 57344);           // 64 tickers (queries)
  int*   kt  = (int*)(smem + 57600);           // 128 tickers (keys)
  float* lse = (float*)(smem + 58112);         // 64 row logits
  float* red = (float*)(smem + 58368);         // 64x4 reduction scratch

  int chunk = blockIdx.x;
  int bh = blockIdx.y;
  int b = bh >> 3, hd = bh & 7;
  int tid = threadIdx.x;
  int lane = tid & 31;
  int wv = tid >> 5;

  if (tid < 64) qt[tid] = sticker[(size_t)bh * C_TT + chunk * 64 + tid];
  if (tid < 128) {
    int cc = (tid < 64) ? chunk : ((chunk + C_NCH - 1) & (C_NCH - 1));
    kt[tid] = sticker[(size_t)bh * C_TT + cc * 64 + (tid & 63)];
  }
  __syncthreads();

  // q tile (unnormalized shared-QK)
  for (int i = tid; i < 64 * 64; i += 256) {
    int row = i >> 6, col = i & 63;
    int pos = qt[row] & (C_T - 1);
    q16[i] = (_Float16)qk[((size_t)(b * C_T + pos)) * C_D + hd * C_DH + col];
  }
  // k tile (L2-normalized) + v tile transposed (dh-major)
  if (tid < 128) {
    int pos = kt[tid] & (C_T - 1);
    const float* kp = qk + ((size_t)(b * C_T + pos)) * C_D + hd * C_DH;
    const float* vp = v  + ((size_t)(b * C_T + pos)) * C_D + hd * C_DH;
    float ss = 0.f;
    for (int c = 0; c < 64; ++c) { float xv = kp[c]; ss += xv * xv; }
    float nrm = sqrtf(ss);
    if (nrm < 1e-6f) nrm = 1e-6f;
    float scl = 1.f / nrm;
    for (int c = 0; c < 64; ++c) k16[tid * 64 + c] = (_Float16)(kp[c] * scl);
    for (int c = 0; c < 64; ++c) vT[c * 128 + tid] = (_Float16)vp[c];
  }
  __syncthreads();

  // dots = (q @ k^T) * dh^-0.5 : 4 Mtiles x 8 Ntiles over 8 waves
  {
    int mt = wv >> 1;
    for (int q4 = 0; q4 < 4; ++q4) {
      int nt = (wv & 1) * 4 + q4;
      v8f acc = {};
      for (int k0 = 0; k0 < 64; k0 += 32) {
        v16h a = frag_a(q16 + mt * 16 * 64 + k0, 64);
        v16h bb = frag_b(k16 + nt * 16 * 64 + k0, 64);
        acc = wmma_f16(a, bb, acc);
      }
      int rr = lane & 15, hi2 = lane >> 4;
#pragma unroll
      for (int vr = 0; vr < 8; ++vr)
        d16[(mt * 16 + vr + hi2 * 8) * 128 + nt * 16 + rr] =
            (_Float16)(acc[vr] * 0.125f);
    }
  }
  __syncthreads();

  // masked logsumexp per row; probs to p16
  {
    int row = tid >> 2;
    int sub = tid & 3;
    int qpos = qt[row] & (C_T - 1);
    int cbase = sub * 32;
    float mx = -3.0e38f;
    for (int c = cbase; c < cbase + 32; ++c) {
      float dv = (float)d16[row * 128 + c];
      if ((kt[c] & (C_T - 1)) == qpos) dv = -5.0e4f;
      mx = fmaxf(mx, dv);
    }
    red[row * 4 + sub] = mx;
    __syncthreads();
    float rmx = fmaxf(fmaxf(red[row * 4 + 0], red[row * 4 + 1]),
                      fmaxf(red[row * 4 + 2], red[row * 4 + 3]));
    __syncthreads();
    float sm = 0.f;
    for (int c = cbase; c < cbase + 32; ++c) {
      float dv = (float)d16[row * 128 + c];
      if ((kt[c] & (C_T - 1)) == qpos) dv = -5.0e4f;
      sm += __expf(dv - rmx);
    }
    red[row * 4 + sub] = sm;
    __syncthreads();
    float rsum = red[row * 4 + 0] + red[row * 4 + 1] + red[row * 4 + 2] +
                 red[row * 4 + 3];
    float lg = rmx + __logf(rsum);
    if (sub == 0) lse[row] = lg;
    for (int c = cbase; c < cbase + 32; ++c) {
      float dv = (float)d16[row * 128 + c];
      if ((kt[c] & (C_T - 1)) == qpos) dv = -5.0e4f;
      p16[row * 128 + c] = (_Float16)__expf(dv - lg);
    }
  }
  __syncthreads();
  if (tid < 64) logits[(size_t)bh * C_TT + qt[tid]] = lse[tid];

  // bo = probs @ v : 4 Mtiles x 4 Ntiles over 8 waves; scatter by ticker
  {
    int mt = wv >> 1;
    for (int q2 = 0; q2 < 2; ++q2) {
      int nt = (wv & 1) * 2 + q2;
      v8f acc = {};
      for (int k0 = 0; k0 < 128; k0 += 32) {
        v16h a = frag_a(p16 + mt * 16 * 128 + k0, 128);
        v16h bb = frag_b(vT + nt * 16 * 128 + k0, 128);
        acc = wmma_f16(a, bb, acc);
      }
      int rr = lane & 15, hi2 = lane >> 4;
#pragma unroll
      for (int vr = 0; vr < 8; ++vr) {
        int row = mt * 16 + vr + hi2 * 8;
        int dcol = nt * 16 + rr;
        int tick = qt[row];
        obuf[((size_t)bh * C_TT + tick) * C_DH + dcol] = acc[vr];
      }
    }
  }
}

// ---------------------------------------------------------------- combine hash rounds
__global__ void combine_kernel(const float* __restrict__ obuf,
                               const float* __restrict__ logits,
                               _Float16* __restrict__ attnh) {
  int pos = blockIdx.x;
  int bh = blockIdx.y;
  int d = threadIdx.x;
  int b = bh >> 3, hd = bh & 7;
  float lg[4];
#pragma unroll
  for (int h = 0; h < 4; ++h)
    lg[h] = logits[(size_t)bh * C_TT + h * C_T + pos];
  float m = fmaxf(fmaxf(lg[0], lg[1]), fmaxf(lg[2], lg[3]));
  float e[4], s = 0.f;
#pragma unroll
  for (int h = 0; h < 4; ++h) { e[h] = __expf(lg[h] - m); s += e[h]; }
  float outv = 0.f;
#pragma unroll
  for (int h = 0; h < 4; ++h)
    outv += obuf[((size_t)bh * C_TT + h * C_T + pos) * C_DH + d] * e[h];
  outv /= s;
  attnh[((size_t)(b * C_T + pos)) * C_D + hd * C_DH + d] = (_Float16)outv;
}

// ---------------------------------------------------------------- layernorm
__global__ void ln_kernel(const float* __restrict__ src,
                          const float* __restrict__ g,
                          const float* __restrict__ bvec,
                          float* __restrict__ dst,
                          _Float16* __restrict__ dst16) {
  __shared__ float red[256];
  __shared__ float stat[2];
  int row = blockIdx.x;
  int tid = threadIdx.x;
  const float* p = src + (size_t)row * C_D;
  float a0 = p[tid], a1 = p[tid + 256];
  red[tid] = a0 + a1;
  __syncthreads();
  for (int s = 128; s > 0; s >>= 1) {
    if (tid < s) red[tid] += red[tid + s];
    __syncthreads();
  }
  if (tid == 0) stat[0] = red[0] / (float)C_D;
  __syncthreads();
  float mean = stat[0];
  float d0 = a0 - mean, d1 = a1 - mean;
  red[tid] = d0 * d0 + d1 * d1;
  __syncthreads();
  for (int s = 128; s > 0; s >>= 1) {
    if (tid < s) red[tid] += red[tid + s];
    __syncthreads();
  }
  if (tid == 0) stat[1] = red[0] / (float)C_D;
  __syncthreads();
  float rstd = rsqrtf(stat[1] + 1e-6f);
  float v0 = d0 * rstd * g[tid] + bvec[tid];
  float v1 = d1 * rstd * g[tid + 256] + bvec[tid + 256];
  dst[(size_t)row * C_D + tid] = v0;
  dst[(size_t)row * C_D + tid + 256] = v1;
  if (dst16) {
    dst16[(size_t)row * C_D + tid] = (_Float16)v0;
    dst16[(size_t)row * C_D + tid + 256] = (_Float16)v1;
  }
}

// ---------------------------------------------------------------- launch
extern "C" void kernel_launch(void* const* d_in, const int* in_sizes, int n_in,
                              void* d_out, int out_size, void* d_ws,
                              size_t ws_size, hipStream_t stream) {
  (void)in_sizes; (void)n_in; (void)out_size; (void)ws_size;
  const float* x     = (const float*)d_in[0];
  const float* rot   = (const float*)d_in[1];
  const float* w_qk  = (const float*)d_in[2];
  const float* w_v   = (const float*)d_in[3];
  const float* w_out = (const float*)d_in[4];
  const float* b_out = (const float*)d_in[5];
  const float* ln1g  = (const float*)d_in[6];
  const float* ln1b  = (const float*)d_in[7];
  const float* w1    = (const float*)d_in[8];
  const float* b1    = (const float*)d_in[9];
  const float* w2    = (const float*)d_in[10];
  const float* b2    = (const float*)d_in[11];
  const float* ln2g  = (const float*)d_in[12];
  const float* ln2b  = (const float*)d_in[13];
  float* out = (float*)d_out;

  char* wsb = (char*)d_ws;
  size_t cur = 0;
  auto alloc = [&](size_t bytes) -> char* {
    char* p = wsb + cur;
    cur += (bytes + 255) & ~(size_t)255;
    return p;
  };
  const size_t SZ_BTD_F = (size_t)C_B * C_T * C_D * 4;   // 32 MB
  const size_t SZ_BTD_H = (size_t)C_B * C_T * C_D * 2;   // 16 MB

  float* qk32   = (float*)alloc(SZ_BTD_F);
  float* v32    = (float*)alloc(SZ_BTD_F);
  float* res1   = (float*)alloc(SZ_BTD_F);               // attn@w_out + b + x
  float* h32    = (float*)alloc(SZ_BTD_F);               // LN1 output
  char*  obufc  = alloc((size_t)C_BH * C_TT * C_DH * 4); // 128 MB, reused for FFN
  float* obuf   = (float*)obufc;
  float* logits = (float*)alloc((size_t)C_BH * C_TT * 4);
  int*   buckets= (int*)alloc((size_t)C_BH * C_TT * 4);
  int*   sticker= (int*)alloc((size_t)C_BH * C_TT * 4);
  int*   offs   = (int*)alloc((size_t)C_BH * 256 * 4);
  _Float16* xh    = (_Float16*)alloc(SZ_BTD_H);
  _Float16* wqkh  = (_Float16*)alloc((size_t)C_D * C_D * 2);
  _Float16* wvh   = (_Float16*)alloc((size_t)C_D * C_D * 2);
  _Float16* wouth = (_Float16*)alloc((size_t)C_D * C_D * 2);
  _Float16* w1h   = (_Float16*)alloc((size_t)C_FF * C_D * 2);
  _Float16* w2h   = (_Float16*)alloc((size_t)C_D * C_FF * 2);
  _Float16* attnh = (_Float16*)alloc(SZ_BTD_H);
  // overlays inside obuf (dead after combine_kernel):
  _Float16* f1h    = (_Float16*)(obufc);                           // 64 MB
  _Float16* h16    = (_Float16*)(obufc + 67108864);                // 16 MB
  float*    ffnres = (float*)(obufc + 67108864 + 16777216);        // 32 MB

  const int NTOK = C_B * C_T;  // 16384

  // f16 staging of inputs/weights
  auto cvt = [&](const float* src, _Float16* dst, int n) {
    cvt_f16_kernel<<<(n + 255) / 256, 256, 0, stream>>>(src, dst, n);
  };
  cvt(x, xh, NTOK * C_D);
  cvt(w_qk, wqkh, C_D * C_D);
  cvt(w_v, wvh, C_D * C_D);
  cvt(w_out, wouth, C_D * C_D);
  cvt(w1, w1h, C_FF * C_D);
  cvt(w2, w2h, C_D * C_FF);

  // qk = x @ w_qk^T ; v = x @ w_v^T   (block macro-tile 128x256)
  gemm_wmma_kernel<<<dim3(NTOK / 128, C_D / 256), 256, 0, stream>>>(
      xh, wqkh, qk32, nullptr, nullptr, nullptr, NTOK, C_D, C_D, 0);
  gemm_wmma_kernel<<<dim3(NTOK / 128, C_D / 256), 256, 0, stream>>>(
      xh, wvh, v32, nullptr, nullptr, nullptr, NTOK, C_D, C_D, 0);

  // LSH buckets, stable counting sort
  hash_kernel<<<dim3(C_T / 64, C_BH), 256, 0, stream>>>(qk32, rot, buckets);
  hist_kernel<<<C_BH, 256, 0, stream>>>(buckets, offs);
  scatter_kernel<<<dim3(256, C_BH), 256, 0, stream>>>(buckets, offs, sticker);

  // chunked attention + round combine
  attn_kernel<<<dim3(C_NCH, C_BH), 256, 0, stream>>>(qk32, v32, sticker, obuf,
                                                     logits);
  combine_kernel<<<dim3(C_T, C_BH), 64, 0, stream>>>(obuf, logits, attnh);

  // out projection + residual, LN1
  gemm_wmma_kernel<<<dim3(NTOK / 128, C_D / 256), 256, 0, stream>>>(
      attnh, wouth, res1, nullptr, b_out, x, NTOK, C_D, C_D, 1);
  ln_kernel<<<NTOK, 256, 0, stream>>>(res1, ln1g, ln1b, h32, h16);

  // FFN: relu(h @ w1^T + b1) @ w2^T + b2 + h, LN2 -> out
  gemm_wmma_kernel<<<dim3(NTOK / 128, C_FF / 256), 256, 0, stream>>>(
      h16, w1h, nullptr, f1h, b1, nullptr, NTOK, C_FF, C_D, 2);
  gemm_wmma_kernel<<<dim3(NTOK / 128, C_D / 256), 256, 0, stream>>>(
      f1h, w2h, ffnres, nullptr, b2, h32, NTOK, C_D, C_FF, 1);
  ln_kernel<<<NTOK, 256, 0, stream>>>(ffnres, ln2g, ln2b, out, nullptr);
}
